// SyntheticAttention_4947802325393
// MI455X (gfx1250) — matrix operand order repaired and checked by reading, not compiled, not run on hardware
//
#include <hip/hip_runtime.h>
#include <hip/hip_bf16.h>
#include <math.h>
#include <stdint.h>

typedef __attribute__((ext_vector_type(16))) __bf16 v16bf;
typedef __attribute__((ext_vector_type(8)))  __bf16 v8bf;
typedef __attribute__((ext_vector_type(4)))  __bf16 v4bf;
typedef __attribute__((ext_vector_type(8)))  float  v8f;
typedef __attribute__((ext_vector_type(4)))  float  v4f;
typedef __attribute__((ext_vector_type(4)))  unsigned int v4u;
typedef __attribute__((ext_vector_type(8)))  int    v8i;
typedef __attribute__((ext_vector_type(4)))  int    v4i;

#define BM 128
#define BN 128
#define BKK 32
#define LDSS 40   // BK + 8 pad -> 80 bytes/row, keeps all fragment reads 16B aligned

union BF16x16 { v16bf v; struct { v8bf lo, hi; } h; };

// ---------------------------------------------------------------------------
// TDM: DMA one [BM x BKK] bf16 tile (row stride lda elements) from global into
// LDS at lds_byte_addr, inserting 4 DWORDs of pad after every 16 DWORDs so the
// LDS row stride is LDSS(=40) bf16 = 80 bytes. Issued by one wave; tracked on
// TENSORcnt. Descriptor packing per CDNA5 ISA.
// ---------------------------------------------------------------------------
__device__ __forceinline__ void tdm_load_tile_bf16(const __bf16* gptr, int lda,
                                                   unsigned lds_byte_addr)
{
    unsigned long long ga = (unsigned long long)(uintptr_t)gptr;

    v4u g0;
    g0[0] = 1u;                                          // count=1, user mode
    g0[1] = lds_byte_addr;                               // lds_addr[31:0]
    g0[2] = (unsigned)(ga & 0xFFFFFFFFu);                // global_addr[31:0]
    g0[3] = (unsigned)((ga >> 32) & 0x1FFFFFFu)          // global_addr[56:32]
          | (2u << 30);                                  // type=2 ("image")

    const unsigned td0 = (unsigned)lda;                  // tensor_dim0 (elems)
    const unsigned td1 = 1u << 20;                       // tensor_dim1 (large)
    const unsigned long long st0 = (unsigned long long)lda; // dim0 stride

    v8i g1;
    g1[0] = (int)((1u << 16)      // data_size = 2 bytes
                | (1u << 20)      // pad_enable
                | (3u << 22)      // pad_interval: 16 DWORDs of data
                | (3u << 25));    // pad_amount:   4 DWORDs of pad
    g1[1] = (int)((td0 & 0xFFFFu) << 16);                               // dim0 lo
    g1[2] = (int)(((td0 >> 16) & 0xFFFFu) | ((td1 & 0xFFFFu) << 16));   // dim0 hi | dim1 lo
    g1[3] = (int)(((td1 >> 16) & 0xFFFFu) | ((unsigned)BKK << 16));     // dim1 hi | tile_dim0
    g1[4] = (int)BM;                                     // tile_dim1=128, tile_dim2=0
    g1[5] = (int)(st0 & 0xFFFFFFFFu);                    // dim0_stride lo
    g1[6] = (int)((st0 >> 32) & 0xFFFFu);                // dim0_stride hi | dim1_stride lo
    g1[7] = 0;

    v4i z4 = {0, 0, 0, 0};
#if __clang_major__ >= 23
    v8i z8 = {0, 0, 0, 0, 0, 0, 0, 0};
    __builtin_amdgcn_tensor_load_to_lds(g0, g1, z4, z4, z8, 0);
#else
    __builtin_amdgcn_tensor_load_to_lds(g0, g1, z4, z4, 0);
#endif
}

// ---------------------------------------------------------------------------
// Tiled bf16 WMMA GEMM: C[M,N] = act(A[M,K] @ B[K,N] + bias)
// 256 threads = 8 waves; wave grid 4(M) x 2(N); each wave: 32x64 = 2x4 WMMA tiles.
// bf16 A-operand tiles are moved by the Tensor Data Mover (double-buffered LDS,
// TDM of tile k+1 overlaps WMMA on tile k). f32 sources are converted to bf16
// through a register-staged LDS path; B is transposed into Bs[n][k] on store.
// ---------------------------------------------------------------------------
template<bool A_BF16, bool B_BF16, bool HAS_BIAS, bool RELU, bool OUT_F32>
__global__ __launch_bounds__(256)
void gemm_wmma_bf16(const void* __restrict__ Ap, const void* __restrict__ Bp,
                    const float* __restrict__ bias, void* __restrict__ Cp,
                    int M, int N, int K, int lda, int ldb, int ldc,
                    long long strideA, long long strideB, long long strideC)
{
    constexpr bool A_TDM = A_BF16;                  // bf16 A tiles go through TDM
    constexpr int  ABUF  = A_TDM ? 2 : 1;

    __shared__ __bf16 As[ABUF][BM][LDSS];
    __shared__ __bf16 Bs[BN][LDSS];                 // stored transposed: Bs[n][k]

    const int tid   = threadIdx.x;
    const int lane  = tid & 31;
    const int wave  = tid >> 5;
    const int wm    = wave >> 1;      // 0..3
    const int wn    = wave & 1;       // 0..1
    const int lmod  = lane & 15;
    const int lhalf = lane >> 4;

    const int n0 = blockIdx.x * BN;
    const int m0 = blockIdx.y * BM;
    const int bz = blockIdx.z;

    const float*  Af = ((const float* )Ap) + (size_t)bz * strideA;
    const __bf16* Ab = ((const __bf16*)Ap) + (size_t)bz * strideA;
    const float*  Bf = ((const float* )Bp) + (size_t)bz * strideB;
    const __bf16* Bb = ((const __bf16*)Bp) + (size_t)bz * strideB;
    float*        Cf = ((float* )Cp) + (size_t)bz * strideC;
    __bf16*       Cb = ((__bf16*)Cp) + (size_t)bz * strideC;

    v8f acc[2][4];
    const v8f vzero = {0.f,0.f,0.f,0.f,0.f,0.f,0.f,0.f};
    #pragma unroll
    for (int mt = 0; mt < 2; ++mt)
        #pragma unroll
        for (int nt = 0; nt < 4; ++nt) acc[mt][nt] = vzero;

    // staging registers for the non-TDM global->LDS pipeline
    v4f  a4[4];
    v4f  b4[4]; v8bf b8[2];

    auto tdmA = [&](int kt) {                       // one wave issues the DMA
        if (tid < 32) {
            const __bf16* gp = Ab + (size_t)m0 * lda + (size_t)kt * BKK;
            tdm_load_tile_bf16(gp, lda, (unsigned)(uintptr_t)(&As[kt & 1][0][0]));
        }
    };
    auto tdmWait = [&]() {
        if (tid < 32) __builtin_amdgcn_s_wait_tensorcnt((short)0);
    };

    auto loadA = [&](int kt) {                      // f32 A: 1024 float4 = 128x32
        const int k0 = kt * BKK;
        #pragma unroll
        for (int i = 0; i < 4; ++i) {
            int f = tid + i * 256;
            int r = f >> 3, kq = (f & 7) << 2;
            a4[i] = *(const v4f*)(Af + (size_t)(m0 + r) * lda + k0 + kq);
        }
    };
    auto stashA = [&]() {
        #pragma unroll
        for (int i = 0; i < 4; ++i) {
            int f = tid + i * 256;
            int r = f >> 3, kq = (f & 7) << 2;
            v4bf t;
            t[0] = (__bf16)a4[i][0]; t[1] = (__bf16)a4[i][1];
            t[2] = (__bf16)a4[i][2]; t[3] = (__bf16)a4[i][3];
            *(v4bf*)&As[0][r][kq] = t;
        }
    };
    auto loadB = [&](int kt) {
        const int k0 = kt * BKK;
        if constexpr (!B_BF16) {
            #pragma unroll
            for (int i = 0; i < 4; ++i) {           // 1024 float4 = 32x128 f32
                int f = tid + i * 256;
                int kk = f >> 5, nq = (f & 31) << 2;
                b4[i] = *(const v4f*)(Bf + (size_t)(k0 + kk) * ldb + n0 + nq);
            }
        } else {
            #pragma unroll
            for (int i = 0; i < 2; ++i) {           // 512 x v8bf = 32x128 bf16
                int g = tid + i * 256;
                int kk = g >> 4, nq = (g & 15) << 3;
                b8[i] = *(const v8bf*)(Bb + (size_t)(k0 + kk) * ldb + n0 + nq);
            }
        }
    };
    auto stashB = [&]() {   // transpose into Bs[n][k]
        if constexpr (!B_BF16) {
            #pragma unroll
            for (int i = 0; i < 4; ++i) {
                int f = tid + i * 256;
                int kk = f >> 5, nq = (f & 31) << 2;
                #pragma unroll
                for (int j = 0; j < 4; ++j) Bs[nq + j][kk] = (__bf16)b4[i][j];
            }
        } else {
            #pragma unroll
            for (int i = 0; i < 2; ++i) {
                int g = tid + i * 256;
                int kk = g >> 4, nq = (g & 15) << 3;
                #pragma unroll
                for (int j = 0; j < 8; ++j) Bs[nq + j][kk] = b8[i][j];
            }
        }
    };

    const int KT = K / BKK;

    // prologue: tile 0 into LDS
    if constexpr (A_TDM) tdmA(0); else loadA(0);
    loadB(0);
    if constexpr (A_TDM) tdmWait(); else stashA();
    stashB();
    __syncthreads();

    for (int kt = 0; kt < KT; ++kt) {
        const bool more = (kt + 1) < KT;
        if (more) {
            if constexpr (A_TDM) tdmA(kt + 1);      // DMA next A tile (other buffer)
            else                 loadA(kt + 1);
            loadB(kt + 1);                          // prefetch next B into regs
        }

        const int abuf = A_TDM ? (kt & 1) : 0;

        // gather fragments (ISA 7.12.2 bf16 layouts)
        BF16x16 afr[2], bfr[4];
        #pragma unroll
        for (int mt = 0; mt < 2; ++mt) {
            int r = wm * 32 + mt * 16 + lmod;
            afr[mt].h.lo = *(const v8bf*)&As[abuf][r][8 * lhalf];
            afr[mt].h.hi = *(const v8bf*)&As[abuf][r][16 + 8 * lhalf];
        }
        #pragma unroll
        for (int nt = 0; nt < 4; ++nt) {
            int c = wn * 64 + nt * 16 + lmod;
            bfr[nt].h.lo = *(const v8bf*)&Bs[c][8 * lhalf];
            bfr[nt].h.hi = *(const v8bf*)&Bs[c][16 + 8 * lhalf];
        }
        #pragma unroll
        for (int mt = 0; mt < 2; ++mt)
            #pragma unroll
            for (int nt = 0; nt < 4; ++nt)
                acc[mt][nt] = __builtin_amdgcn_wmma_f32_16x16x32_bf16(
                    false, afr[mt].v, false, bfr[nt].v,
                    (short)0, acc[mt][nt], false, false);

        __syncthreads();                            // done reading As[abuf], Bs
        if (more) {
            if constexpr (!A_TDM) stashA();
            stashB();
            if constexpr (A_TDM) tdmWait();         // next A tile landed in LDS
            __syncthreads();
        }
    }

    // epilogue: D layout -> lane holds col=lmod, rows r + 8*lhalf
    #pragma unroll
    for (int mt = 0; mt < 2; ++mt) {
        #pragma unroll
        for (int nt = 0; nt < 4; ++nt) {
            const int col = n0 + wn * 64 + nt * 16 + lmod;
            float bv = 0.f;
            if constexpr (HAS_BIAS) bv = bias[col];
            #pragma unroll
            for (int r = 0; r < 8; ++r) {
                const int row = m0 + wm * 32 + mt * 16 + 8 * lhalf + r;
                float v = acc[mt][nt][r] + bv;
                if constexpr (RELU) v = fmaxf(v, 0.f);
                if constexpr (OUT_F32) Cf[(size_t)row * ldc + col] = v;
                else                   Cb[(size_t)row * ldc + col] = (__bf16)v;
            }
        }
    }
}

// ---------------------------------------------------------------------------
// Row softmax over 2048 columns, in place on the f32 attention output, plus a
// bf16 copy for the following WMMA GEMM. One 256-thread block per row.
// ---------------------------------------------------------------------------
__global__ __launch_bounds__(256)
void softmax_rows(float* __restrict__ attnF, __bf16* __restrict__ attnB,
                  const int* __restrict__ mask, int Lrow)
{
    const int row = blockIdx.x;            // 0..B*L-1
    const int b   = row / Lrow;
    const int tid = threadIdx.x;
    float* p = attnF + (size_t)row * Lrow;
    const int* mrow = mask + (size_t)b * Lrow;

    __shared__ float red[256];

    float lv[8];
    float mx = -INFINITY;
    #pragma unroll
    for (int i = 0; i < 8; ++i) {
        int c = tid + i * 256;
        float v = p[c];
        if (mrow[c] == 0) v = -INFINITY;
        lv[i] = v;
        mx = fmaxf(mx, v);
    }
    red[tid] = mx; __syncthreads();
    for (int s = 128; s > 0; s >>= 1) {
        if (tid < s) red[tid] = fmaxf(red[tid], red[tid + s]);
        __syncthreads();
    }
    mx = red[0]; __syncthreads();

    float sum = 0.f;
    #pragma unroll
    for (int i = 0; i < 8; ++i) {
        float e = (lv[i] == -INFINITY) ? 0.f : __expf(lv[i] - mx);
        lv[i] = e;
        sum += e;
    }
    red[tid] = sum; __syncthreads();
    for (int s = 128; s > 0; s >>= 1) {
        if (tid < s) red[tid] += red[tid + s];
        __syncthreads();
    }
    sum = red[0];
    const float inv = (sum > 0.f) ? (1.f / sum) : 0.f;

    #pragma unroll
    for (int i = 0; i < 8; ++i) {
        int c = tid + i * 256;
        float a = lv[i] * inv;
        p[c] = a;
        attnB[(size_t)row * Lrow + c] = (__bf16)a;
    }
}

// ---------------------------------------------------------------------------
extern "C" void kernel_launch(void* const* d_in, const int* in_sizes, int n_in,
                              void* d_out, int out_size, void* d_ws, size_t ws_size,
                              hipStream_t stream)
{
    (void)in_sizes; (void)n_in; (void)out_size; (void)ws_size;
    const int Bb = 4, L = 2048, D = 1024, MAXL = 4096;
    const int M = Bb * L;                 // 8192

    const float* x    = (const float*)d_in[0];
    const int*   mask = (const int*  )d_in[1];
    const float* W1   = (const float*)d_in[2];
    const float* b1   = (const float*)d_in[3];
    const float* W2   = (const float*)d_in[4];
    const float* b2   = (const float*)d_in[5];
    const float* D2L  = (const float*)d_in[6];

    float* outP  = (float*)d_out;                         // [B,L,D]
    float* attnF = outP + (size_t)Bb * L * D;             // [B,L,L]

    // workspace layout (bf16): h[8192,1024] | kv[8192,2048] | attn[8192,2048]
    __bf16* h   = (__bf16*)d_ws;
    __bf16* kv  = h  + (size_t)M * D;
    __bf16* atb = kv + (size_t)M * 2 * D;

    dim3 blk(256);

    // 1) h = relu(x @ W1 + b1)            f32 x f32 -> bf16 (staged path)
    gemm_wmma_bf16<false, false, true, true, false>
        <<<dim3(D / BN, M / BM, 1), blk, 0, stream>>>(
            x, W1, b1, h, M, D, D, D, D, D, 0, 0, 0);

    // 2) kv = h @ W2 + b2                 bf16(TDM) x f32 -> bf16
    gemm_wmma_bf16<true, false, true, false, false>
        <<<dim3(2 * D / BN, M / BM, 1), blk, 0, stream>>>(
            h, W2, b2, kv, M, 2 * D, D, D, 2 * D, 2 * D, 0, 0, 0);

    // 3) scores = k @ D_to_L[:, :L]       bf16(TDM) x f32 -> f32 (attention out)
    gemm_wmma_bf16<true, false, false, false, true>
        <<<dim3(L / BN, M / BM, 1), blk, 0, stream>>>(
            kv, D2L, nullptr, attnF, M, L, D, 2 * D, MAXL, L, 0, 0, 0);

    // 4) attention = softmax(mask(scores)) in place; bf16 copy for next GEMM
    softmax_rows<<<dim3(M), blk, 0, stream>>>(attnF, atb, mask, L);

    // 5) output = attention @ value       bf16(TDM) x bf16 -> f32, batched over B
    gemm_wmma_bf16<true, true, false, false, true>
        <<<dim3(D / BN, L / BM, Bb), blk, 0, stream>>>(
            atb, kv + D, nullptr, outP, L, D, L, L, 2 * D, D,
            (long long)L * L, (long long)L * 2 * D, (long long)L * D);
}
